// MHN_WITH_1_HIDDEN_LAYER_2147483648713
// MI455X (gfx1250) — matrix operand order repaired and checked by reading, not compiled, not run on hardware
//
#include <hip/hip_runtime.h>
#include <cstdint>
#include <cstddef>

typedef __attribute__((ext_vector_type(16))) __bf16 v16bf;
typedef __attribute__((ext_vector_type(8)))  __bf16 v8bf;
typedef __attribute__((ext_vector_type(8)))  float  v8f;

static constexpr int M_TOT  = 65536;
static constexpr int D_DIM  = 256;
static constexpr int B_TOT  = 512;
static constexpr int NB     = 64;              // columns per block
static constexpr int CT     = B_TOT / NB;      // 8 column tiles
static constexpr int MS     = 32;              // m-splits
static constexpr int MCHUNK = M_TOT / MS;      // 2048 memories per block
static constexpr int SLAB   = 128;
static constexpr int NSLAB  = MCHUNK / SLAB;   // 16
static constexpr int VSTR   = 264;             // LDS row stride (bf16 elems) for v tiles
static constexpr int WSTR   = 136;             // LDS row stride (bf16 elems) for w slab
static constexpr float ALPHA = 0.5f;

// ---------- helpers ----------
__device__ __forceinline__ unsigned short f2bf(float f) {
  union { float f; unsigned u; } c; c.f = f;
  unsigned r = c.u + 0x7FFFu + ((c.u >> 16) & 1u);   // round-to-nearest-even
  return (unsigned short)(r >> 16);
}
__device__ __forceinline__ float bf2f(unsigned short h) {
  union { unsigned u; float f; } c; c.u = ((unsigned)h) << 16;
  return c.f;
}
// A-matrix fragment (16-bit 16x32): per-lane K = 16 contiguous values
__device__ __forceinline__ v16bf mkfragA(const unsigned short* p) {
  v16bf r;
  ((v8bf*)&r)[0] = *(const v8bf*)(p);
  ((v8bf*)&r)[1] = *(const v8bf*)(p + 8);
  return r;
}
// B-matrix fragment (16-bit 32x16): per-lane K = 16 contiguous values
__device__ __forceinline__ v16bf mkfragB(const unsigned short* p) {
  v16bf r;
  ((v8bf*)&r)[0] = *(const v8bf*)(p);
  ((v8bf*)&r)[1] = *(const v8bf*)(p + 8);
  return r;
}
__device__ __forceinline__ v8f wmma_bf16(v16bf a, v16bf b, v8f c) {
  return __builtin_amdgcn_wmma_f32_16x16x32_bf16(false, a, false, b, (short)0, c,
                                                 false, false);
}

// ---------- prep 1: fp32 memories -> bf16 hi/lo + row squared norms ----------
__global__ __launch_bounds__(256) void prep_rows(const float* __restrict__ mem,
                                                 unsigned short* __restrict__ hi,
                                                 unsigned short* __restrict__ lo,
                                                 float* __restrict__ msq) {
  const int lane = threadIdx.x & 31;
  const int gw   = (blockIdx.x * blockDim.x + threadIdx.x) >> 5;
  const int nw   = (gridDim.x * blockDim.x) >> 5;
  for (int m = gw; m < M_TOT; m += nw) {
    const float* row = mem + (size_t)m * D_DIM;
    float s = 0.f;
    #pragma unroll
    for (int c = 0; c < D_DIM; c += 32) {
      float x = row[c + lane];
      unsigned short h = f2bf(x);
      hi[(size_t)m * D_DIM + c + lane] = h;
      lo[(size_t)m * D_DIM + c + lane] = f2bf(x - bf2f(h));
      s = fmaf(x, x, s);
    }
    #pragma unroll
    for (int o = 16; o >= 1; o >>= 1) s += __shfl_xor(s, o, 32);
    if (lane == 0) msq[m] = s;
  }
}

// ---------- prep 2: transposed bf16-hi copy memT[d][m] ----------
__global__ __launch_bounds__(256) void prep_T(const float* __restrict__ mem,
                                              unsigned short* __restrict__ memT) {
  __shared__ unsigned short tile[64][65];
  const int m0 = blockIdx.x * 64, d0 = blockIdx.y * 64;
  const int j = threadIdx.x & 63, i0 = threadIdx.x >> 6;
  #pragma unroll
  for (int r = 0; r < 64; r += 4) {
    int i = r + i0;
    tile[i][j] = f2bf(mem[(size_t)(m0 + i) * D_DIM + d0 + j]);
  }
  __syncthreads();
  #pragma unroll
  for (int r = 0; r < 64; r += 4) {
    int jj = r + i0;
    memT[(size_t)(d0 + jj) * M_TOT + m0 + j] = tile[j][jj];
  }
}

// ---------- main: logits(max) -> exp -> numerator, per (col-tile, m-split) ----------
__global__ __launch_bounds__(256) void mhn_main(const float* __restrict__ v,
                                                const unsigned short* __restrict__ mhi,
                                                const unsigned short* __restrict__ mlo,
                                                const unsigned short* __restrict__ mT,
                                                const float* __restrict__ msq,
                                                float* __restrict__ pnum,
                                                float* __restrict__ pmax,
                                                float* __restrict__ pden) {
  __shared__ __align__(16) unsigned short vhiL[NB * VSTR];
  __shared__ __align__(16) unsigned short vloL[NB * VSTR];
  __shared__ __align__(16) unsigned short wbuf[NB * WSTR];
  __shared__ float maxbuf[NB];
  __shared__ float scratch[8][NB];

  const int ct = blockIdx.x, msid = blockIdx.y;
  const int b0 = ct * NB;
  const int m0 = msid * MCHUNK;
  const int tid = threadIdx.x, wave = tid >> 5, lane = tid & 31;
  const int l16 = lane & 15, hh = lane >> 4;

  const v8f zero8 = {};

  // load v tile -> hi/lo bf16 in LDS
  for (int idx = tid; idx < NB * D_DIM; idx += 256) {
    int bi = idx >> 8, dd = idx & 255;
    float x = v[(size_t)(b0 + bi) * D_DIM + dd];
    unsigned short h = f2bf(x);
    vhiL[bi * VSTR + dd] = h;
    vloL[bi * VSTR + dd] = f2bf(x - bf2f(h));
  }
  __syncthreads();

  // ---- pass 1: per-column max of (cross - 0.5*||mem||^2), hi*hi estimate ----
  v8f maxacc[4];
  #pragma unroll
  for (int bg = 0; bg < 4; ++bg) maxacc[bg] = zero8 + (-3.0e38f);

  for (int p = 0; p < 8; ++p) {          // this wave's 8 tile-pairs (16 m-tiles)
    const int t0 = wave * 16 + p * 2;
    const int mA = m0 + t0 * 16 + l16;
    const int mB = mA + 16;
    const unsigned short* rA = mhi + (size_t)mA * D_DIM;
    const unsigned short* rB = mhi + (size_t)mB * D_DIM;
    v8f acc0[4], acc1[4];
    #pragma unroll
    for (int bg = 0; bg < 4; ++bg) { acc0[bg] = zero8; acc1[bg] = zero8; }
    #pragma unroll
    for (int k = 0; k < 8; ++k) {
      const int offB = k * 32 + hh * 16;   // B frag: contiguous K
      const int offA = k * 32 + hh * 16;   // A frag: contiguous K
      v16bf bA = mkfragB(rA + offB);
      v16bf bB = mkfragB(rB + offB);
      #pragma unroll
      for (int bg = 0; bg < 4; ++bg) {
        v16bf a = mkfragA(vhiL + (bg * 16 + l16) * VSTR + offA);
        acc0[bg] = wmma_bf16(a, bA, acc0[bg]);
        acc1[bg] = wmma_bf16(a, bB, acc1[bg]);
      }
    }
    const float qA = 0.5f * msq[mA];
    const float qB = 0.5f * msq[mB];
    #pragma unroll
    for (int bg = 0; bg < 4; ++bg)
      #pragma unroll
      for (int j = 0; j < 8; ++j)
        maxacc[bg][j] = fmaxf(maxacc[bg][j],
                              fmaxf(acc0[bg][j] - qA, acc1[bg][j] - qB));
  }
  // reduce across 16-lane column groups, then across waves via LDS
  #pragma unroll
  for (int bg = 0; bg < 4; ++bg)
    #pragma unroll
    for (int j = 0; j < 8; ++j) {
      float x = maxacc[bg][j];
      #pragma unroll
      for (int o = 8; o >= 1; o >>= 1) x = fmaxf(x, __shfl_xor(x, o, 16));
      maxacc[bg][j] = x;
    }
  if (l16 == 0) {
    #pragma unroll
    for (int bg = 0; bg < 4; ++bg)
      #pragma unroll
      for (int j = 0; j < 8; ++j)
        scratch[wave][bg * 16 + hh * 8 + j] = maxacc[bg][j];
  }
  __syncthreads();
  if (tid < NB) {
    float mx = -3.0e38f;
    #pragma unroll
    for (int w = 0; w < 8; ++w) mx = fmaxf(mx, scratch[w][tid]);
    maxbuf[tid] = mx;
  }
  __syncthreads();

  // ---- pass 2: split-precision logits -> exp -> w ; GEMM2 accumulates numer ----
  v8f num[4][2];
  v8f dacc[4];
  #pragma unroll
  for (int bg = 0; bg < 4; ++bg) {
    num[bg][0] = zero8; num[bg][1] = zero8; dacc[bg] = zero8;
  }

  for (int slab = 0; slab < NSLAB; ++slab) {
    // (a) this wave computes weights for its m-tile of the slab (all 4 b-groups)
    const int tt = slab * 8 + wave;
    const int mA = m0 + tt * 16 + l16;
    const unsigned short* rH = mhi + (size_t)mA * D_DIM;
    const unsigned short* rL = mlo + (size_t)mA * D_DIM;
    v8f acc[4];
    #pragma unroll
    for (int bg = 0; bg < 4; ++bg) acc[bg] = zero8;
    #pragma unroll
    for (int k = 0; k < 8; ++k) {
      const int offB = k * 32 + hh * 16;
      const int offA = k * 32 + hh * 16;
      v16bf bh = mkfragB(rH + offB);
      v16bf bl = mkfragB(rL + offB);
      #pragma unroll
      for (int bg = 0; bg < 4; ++bg) {
        v16bf ah = mkfragA(vhiL + (bg * 16 + l16) * VSTR + offA);
        v16bf al = mkfragA(vloL + (bg * 16 + l16) * VSTR + offA);
        acc[bg] = wmma_bf16(ah, bh, acc[bg]);   // hi*hi
        acc[bg] = wmma_bf16(ah, bl, acc[bg]);   // hi*lo
        acc[bg] = wmma_bf16(al, bh, acc[bg]);   // lo*hi
      }
    }
    const float q = 0.5f * msq[mA];
    const int mloc = wave * 16 + l16;           // slab-local m column
    #pragma unroll
    for (int bg = 0; bg < 4; ++bg)
      #pragma unroll
      for (int j = 0; j < 8; ++j) {
        const int row = bg * 16 + hh * 8 + j;
        float wv = __expf(acc[bg][j] - q - maxbuf[row]);
        unsigned short wb = f2bf(wv);
        dacc[bg][j] += bf2f(wb);                // denom from same rounded w
        wbuf[row * WSTR + mloc] = wb;
      }
    __syncthreads();

    // (b) GEMM2: num[b][d] += w[b][m] * mem[m][d] over this slab
    const int dbase = wave * 32 + l16;          // d column of this wave's dt0
    #pragma unroll
    for (int ms2 = 0; ms2 < 4; ++ms2) {
      const int mo = slab * SLAB + ms2 * 32 + hh * 16;
      const unsigned short* t0p = mT + (size_t)dbase * M_TOT + m0 + mo;
      const unsigned short* t1p = t0p + (size_t)16 * M_TOT;
      v16bf bm0 = mkfragB(t0p);
      v16bf bm1 = mkfragB(t1p);
      const int offA = ms2 * 32 + hh * 16;
      #pragma unroll
      for (int bg = 0; bg < 4; ++bg) {
        v16bf aw = mkfragA(wbuf + (bg * 16 + l16) * WSTR + offA);
        num[bg][0] = wmma_bf16(aw, bm0, num[bg][0]);
        num[bg][1] = wmma_bf16(aw, bm1, num[bg][1]);
      }
    }
    __syncthreads();
  }

  // ---- write partials ----
  #pragma unroll
  for (int bg = 0; bg < 4; ++bg)
    #pragma unroll
    for (int j = 0; j < 8; ++j) {
      float s = dacc[bg][j];
      #pragma unroll
      for (int o = 8; o >= 1; o >>= 1) s += __shfl_xor(s, o, 16);
      dacc[bg][j] = s;
    }
  if (l16 == 0) {
    #pragma unroll
    for (int bg = 0; bg < 4; ++bg)
      #pragma unroll
      for (int j = 0; j < 8; ++j)
        scratch[wave][bg * 16 + hh * 8 + j] = dacc[bg][j];
  }
  __syncthreads();
  const size_t pbase = ((size_t)ct * MS + msid) * NB;
  if (tid < NB) {
    float s = 0.f;
    #pragma unroll
    for (int w = 0; w < 8; ++w) s += scratch[w][tid];
    pden[pbase + tid] = s;
    pmax[pbase + tid] = maxbuf[tid];
  }
  #pragma unroll
  for (int bg = 0; bg < 4; ++bg)
    #pragma unroll
    for (int dts = 0; dts < 2; ++dts)
      #pragma unroll
      for (int j = 0; j < 8; ++j) {
        const int row = bg * 16 + hh * 8 + j;
        const int dcol = (wave * 2 + dts) * 16 + l16;
        pnum[(pbase + row) * D_DIM + dcol] = num[bg][dts][j];
      }
}

// ---------- finalize: merge m-splits (flash merge) and apply update ----------
__global__ __launch_bounds__(256) void mhn_final(const float* __restrict__ v,
                                                 const float* __restrict__ mask,
                                                 const float* __restrict__ pnum,
                                                 const float* __restrict__ pmax,
                                                 const float* __restrict__ pden,
                                                 float* __restrict__ out) {
  __shared__ float smax[MS], sden[MS];
  const int b = blockIdx.x;
  const int ct = b >> 6, bl = b & 63;
  const int d = threadIdx.x;
  if (d < MS) {
    size_t ix = ((size_t)ct * MS + d) * NB + bl;
    smax[d] = pmax[ix];
    sden[d] = pden[ix];
  }
  __syncthreads();
  float g = -3.0e38f;
  #pragma unroll
  for (int i = 0; i < MS; ++i) g = fmaxf(g, smax[i]);
  float den = 0.f, numd = 0.f;
  for (int i = 0; i < MS; ++i) {
    float sc = __expf(smax[i] - g);
    den += sc * sden[i];
    numd += sc * pnum[(((size_t)ct * MS + i) * NB + bl) * D_DIM + d];
  }
  const float vv = v[(size_t)b * D_DIM + d];
  const float mk = mask[(size_t)b * D_DIM + d];
  const float upd = numd / den - vv;
  out[(size_t)b * D_DIM + d] = vv + ALPHA * upd * mk;
}

// ---------- launch ----------
extern "C" void kernel_launch(void* const* d_in, const int* in_sizes, int n_in,
                              void* d_out, int out_size, void* d_ws, size_t ws_size,
                              hipStream_t stream) {
  const float* v    = (const float*)d_in[0];
  const float* mask = (const float*)d_in[1];
  const float* mem  = (const float*)d_in[2];
  float* out = (float*)d_out;

  char* ws = (char*)d_ws;
  // workspace layout (total ~118 MB; everything fits in 192 MB L2)
  constexpr size_t SZ_HI  = (size_t)M_TOT * D_DIM * 2;   // 32 MB
  constexpr size_t SZ_LO  = SZ_HI;                       // 32 MB
  constexpr size_t SZ_T   = SZ_HI;                       // 32 MB
  constexpr size_t SZ_MSQ = (size_t)M_TOT * 4;           // 256 KB
  constexpr size_t SZ_PN  = (size_t)CT * MS * NB * D_DIM * 4;  // 16 MB
  constexpr size_t SZ_PS  = (size_t)CT * MS * NB * 4;    // 64 KB
  unsigned short* mhi = (unsigned short*)(ws);
  unsigned short* mlo = (unsigned short*)(ws + SZ_HI);
  unsigned short* mT  = (unsigned short*)(ws + SZ_HI + SZ_LO);
  float* msq  = (float*)(ws + SZ_HI + SZ_LO + SZ_T);
  float* pnum = (float*)(ws + SZ_HI + SZ_LO + SZ_T + SZ_MSQ);
  float* pmax = (float*)(ws + SZ_HI + SZ_LO + SZ_T + SZ_MSQ + SZ_PN);
  float* pden = (float*)(ws + SZ_HI + SZ_LO + SZ_T + SZ_MSQ + SZ_PN + SZ_PS);

  prep_rows<<<1024, 256, 0, stream>>>(mem, mhi, mlo, msq);
  prep_T<<<dim3(M_TOT / 64, D_DIM / 64), 256, 0, stream>>>(mem, mT);
  mhn_main<<<dim3(CT, MS), 256, 0, stream>>>(v, mhi, mlo, mT, msq, pnum, pmax, pden);
  mhn_final<<<B_TOT, 256, 0, stream>>>(v, mask, pnum, pmax, pden, out);
}
